// HiGraphLatentEncoderCond_63221918597341
// MI455X (gfx1250) — matrix operand order repaired and checked by reading, not compiled, not run on hardware
//
#include <hip/hip_runtime.h>

// ---------------------------------------------------------------------------
// CDNA5 (gfx1250) implementation of the HiGraph latent encoder.
// All MLPs run through v_wmma_f32_16x16x32_bf16 (wave32, 16x16 tiles).
// Round 2 changes:
//  * 32 rows per wave (two 16-row subtiles share every weight B-fragment)
//    -> halves L2 weight-fragment traffic, 2 WMMAs per 32B/lane loaded.
//  * software-pipelined B-fragment prefetch (bnext) so weight loads overlap
//    the WMMAs instead of load -> s_wait_loadcnt 0 -> wmma serialization.
// Weights are pre-packed once per launch into WMMA B-fragment order
// (L2-resident: 192MB L2 vs ~3.4MB of packed weights).
// Workspace requirement: ~140 MB.
// ---------------------------------------------------------------------------

typedef __attribute__((ext_vector_type(16))) __bf16        v16bf;
typedef __attribute__((ext_vector_type(8)))  float         v8f;
typedef __attribute__((ext_vector_type(4)))  unsigned int  u32x4;
typedef __attribute__((ext_vector_type(2)))  unsigned int  u32x2;
typedef __attribute__((ext_vector_type(4)))  float         f32x4;

struct U32x8 { u32x4 lo, hi; };
static_assert(sizeof(U32x8) == 32, "frag size");
static_assert(sizeof(v16bf) == 32, "v16bf size");

#define LDPAD 8   // ushort pad per LDS row -> +4 dwords, kills bank conflicts

__device__ __forceinline__ unsigned short f2bfbits(float f) {
  unsigned u = __builtin_bit_cast(unsigned, f);
  u += 0x7FFFu + ((u >> 16) & 1u);           // round-to-nearest-even
  return (unsigned short)(u >> 16);
}

__device__ __forceinline__ f32x4 ld4(const float* p) { return *(const f32x4*)p; }

__device__ __forceinline__ void st4bf(unsigned short* p, f32x4 v) {
  unsigned a = f2bfbits(v.x), b = f2bfbits(v.y), c = f2bfbits(v.z), d = f2bfbits(v.w);
  u32x2 u; u.x = a | (b << 16); u.y = c | (d << 16);
  *(u32x2*)p = u;
}

// Build a v16bf fragment from two 16B runs (A: hiOff=16 K-elems, B: hiOff=8).
__device__ __forceinline__ v16bf ld_frag(const unsigned short* p, int hiOffElems) {
  U32x8 u;
  u.lo = *(const u32x4*)p;
  u.hi = *(const u32x4*)(p + hiOffElems);
  return __builtin_bit_cast(v16bf, u);
}

__device__ __forceinline__ v8f wmma_bf16(v16bf a, v16bf b, v8f c) {
  return __builtin_amdgcn_wmma_f32_16x16x32_bf16(false, a, false, b, (short)0, c,
                                                 false, false);
}

// ---------------------------------------------------------------------------
// Wave-level 2-layer MLP on a 32-row tile (two 16-row WMMA subtiles):
//   h1 = silu(X @ W1 + b1)  (KIN -> 128), written back to LDS in A-layout
//   Y  = h1 @ W2 + b2       (128 -> 128), returned in C-layout acc2[2][8]
// X: bf16 in LDS, 32 rows, pitch KIN+LDPAD.
// W1p/W2p are packed B-fragments: [(kt*8+nt)*32 + lane][16] bf16; each
// fragment is consumed by BOTH subtiles (2 WMMAs per fragment load) and is
// prefetched one iteration ahead.
// ---------------------------------------------------------------------------
template<int KIN>
__device__ __forceinline__ void wave_mlp2_r32(unsigned short* xs,
                                              const unsigned short* __restrict__ W1p,
                                              const unsigned short* __restrict__ W2p,
                                              const float* __restrict__ b1,
                                              const float* __restrict__ b2,
                                              v8f acc2[2][8], int lane)
{
  constexpr int XP  = KIN + LDPAD;
  constexpr int KT1 = KIN / 32;
  constexpr int HP  = 128 + LDPAD;
  const int lg = lane >> 4;      // half-wave group
  const int ln = lane & 15;

  v8f acc[2][8];
#pragma unroll
  for (int st = 0; st < 2; ++st)
#pragma unroll
    for (int nt = 0; nt < 8; ++nt)
      acc[st][nt] = v8f{0.f,0.f,0.f,0.f,0.f,0.f,0.f,0.f};

  // ---- GEMM1: 32 x KIN @ KIN x 128, B prefetched one step ahead ---------
  v16bf bnext = ld_frag(W1p + (size_t)lane * 16, 8);   // fragment 0
#pragma unroll
  for (int kt = 0; kt < KT1; ++kt) {
    // A fragments: lane holds row M=ln (+16 for subtile 1);
    // K = kt*32 + lg*8 + {0..7, 16..23}
    const unsigned short* xr0 = xs + (unsigned)ln * XP        + kt * 32 + lg * 8;
    const unsigned short* xr1 = xs + (unsigned)(16 + ln) * XP + kt * 32 + lg * 8;
    v16bf a0 = ld_frag(xr0, 16);
    v16bf a1 = ld_frag(xr1, 16);
#pragma unroll
    for (int nt = 0; nt < 8; ++nt) {
      v16bf b = bnext;
      const int ni = kt * 8 + nt + 1;
      if (ni < KT1 * 8)
        bnext = ld_frag(W1p + (size_t)ni * 512 + (size_t)lane * 16, 8);
      acc[0][nt] = wmma_bf16(a0, b, acc[0][nt]);
      acc[1][nt] = wmma_bf16(a1, b, acc[1][nt]);
    }
  }

  // Prefetch first W2 fragment; its latency hides behind the SiLU epilogue.
  v16bf b2next = ld_frag(W2p + (size_t)lane * 16, 8);

  // ---- bias + SiLU, write h1 back to LDS in A-layout (pitch HP) ---------
#pragma unroll
  for (int st = 0; st < 2; ++st) {
#pragma unroll
    for (int nt = 0; nt < 8; ++nt) {
      float bb = b1[nt * 16 + ln];
#pragma unroll
      for (int r = 0; r < 8; ++r) {
        float v = acc[st][nt][r] + bb;       // row M = st*16 + r + lg*8
        float h = v / (1.f + __expf(-v));    // silu
        xs[(unsigned)(st * 16 + r + lg * 8) * HP + nt * 16 + ln] = f2bfbits(h);
      }
    }
  }

  // ---- GEMM2: 32 x 128 @ 128 x 128 --------------------------------------
#pragma unroll
  for (int st = 0; st < 2; ++st)
#pragma unroll
    for (int nt = 0; nt < 8; ++nt)
      acc2[st][nt] = v8f{0.f,0.f,0.f,0.f,0.f,0.f,0.f,0.f};
#pragma unroll
  for (int kt = 0; kt < 4; ++kt) {
    const unsigned short* xr0 = xs + (unsigned)ln * HP        + kt * 32 + lg * 8;
    const unsigned short* xr1 = xs + (unsigned)(16 + ln) * HP + kt * 32 + lg * 8;
    v16bf a0 = ld_frag(xr0, 16);
    v16bf a1 = ld_frag(xr1, 16);
#pragma unroll
    for (int nt = 0; nt < 8; ++nt) {
      v16bf b = b2next;
      const int ni = kt * 8 + nt + 1;
      if (ni < 32)
        b2next = ld_frag(W2p + (size_t)ni * 512 + (size_t)lane * 16, 8);
      acc2[0][nt] = wmma_bf16(a0, b, acc2[0][nt]);
      acc2[1][nt] = wmma_bf16(a1, b, acc2[1][nt]);
    }
  }
#pragma unroll
  for (int st = 0; st < 2; ++st)
#pragma unroll
    for (int nt = 0; nt < 8; ++nt) {
      float bb = b2[nt * 16 + ln];
#pragma unroll
      for (int r = 0; r < 8; ++r) acc2[st][nt][r] += bb;
    }
}

// LayerNorm over the 128 columns of each row held in C-layout acc2.
// Row m lives at vgpr (m&7) across the 16 lanes of one half-wave -> reduce
// with shfl_xor masks 1,2,4,8 (stays inside the 16-lane group).
__device__ __forceinline__ void wave_ln(v8f acc2[8],
                                        const float* __restrict__ ls,
                                        const float* __restrict__ lb, int lane)
{
  const int ln = lane & 15;
  float lsv[8], lbv[8];
#pragma unroll
  for (int nt = 0; nt < 8; ++nt) { lsv[nt] = ls[nt*16+ln]; lbv[nt] = lb[nt*16+ln]; }
#pragma unroll
  for (int r = 0; r < 8; ++r) {
    float s = 0.f, q = 0.f;
#pragma unroll
    for (int nt = 0; nt < 8; ++nt) { float v = acc2[nt][r]; s += v; q += v * v; }
#pragma unroll
    for (int off = 1; off < 16; off <<= 1) {
      s += __shfl_xor(s, off, 32);
      q += __shfl_xor(q, off, 32);
    }
    float mean = s * (1.f / 128.f);
    float var  = q * (1.f / 128.f) - mean * mean;
    float rstd = rsqrtf(var + 1e-5f);
#pragma unroll
    for (int nt = 0; nt < 8; ++nt)
      acc2[nt][r] = (acc2[nt][r] - mean) * rstd * lsv[nt] + lbv[nt];
  }
}

// ---------------------------------------------------------------------------
// Edge MLP kernel: one wave per 32 edges (2 waves / 64-thread block);
// gathers [feat, send[src], recv[dst]], runs 384->128->128 MLP+LN,
// scatter-adds e_up into aggr[dst], optionally edge_out = feat + e_up.
// ---------------------------------------------------------------------------
__global__ __launch_bounds__(64)
void edge_mlp_kernel(const float* __restrict__ efeat,
                     const int*   __restrict__ src,
                     const int*   __restrict__ dst,
                     const float* __restrict__ send,
                     const float* __restrict__ recv,
                     const unsigned short* __restrict__ W1p,
                     const unsigned short* __restrict__ W2p,
                     const float* __restrict__ b1, const float* __restrict__ b2,
                     const float* __restrict__ ls, const float* __restrict__ lb,
                     float* __restrict__ aggr,
                     float* __restrict__ eout,
                     int E)
{
  constexpr int KIN = 384, XP = KIN + LDPAD;
  __shared__ unsigned short smem[2 * 32 * XP];          // 50176 B
  const int wid  = threadIdx.x >> 5;
  const int lane = threadIdx.x & 31;
  const int lg = lane >> 4, ln = lane & 15;
  unsigned short* xs = smem + (unsigned)wid * 32 * XP;
  const int e0 = (blockIdx.x * 2 + wid) * 32;
  if (e0 >= E) return;                                   // wave-uniform

  // gather + fp32->bf16 into LDS (wave-private tile, DS ops are in-order)
  const int c = lane * 4;
#pragma unroll 1
  for (int m = 0; m < 32; ++m) {
    int e = e0 + m;
    int s = src[e], d = dst[e];
    st4bf(xs + m * XP +   0 + c, ld4(efeat + (size_t)e * 128 + c));
    st4bf(xs + m * XP + 128 + c, ld4(send  + (size_t)s * 128 + c));
    st4bf(xs + m * XP + 256 + c, ld4(recv  + (size_t)d * 128 + c));
  }

  v8f acc2[2][8];
  wave_mlp2_r32<384>(xs, W1p, W2p, b1, b2, acc2, lane);
  wave_ln(acc2[0], ls, lb, lane);
  wave_ln(acc2[1], ls, lb, lane);

  // scatter e_up into aggr (segment_sum) + optional residual edge output
#pragma unroll
  for (int st = 0; st < 2; ++st) {
#pragma unroll
    for (int r = 0; r < 8; ++r) {
      int e = e0 + st * 16 + r + lg * 8;
      int d = dst[e];
      float* arow = aggr + (size_t)d * 128;
#pragma unroll
      for (int nt = 0; nt < 8; ++nt) {
        int col = nt * 16 + ln;
        float v = acc2[st][nt][r];
        atomicAdd(arow + col, v);
        if (eout) eout[(size_t)e * 128 + col] = efeat[(size_t)e * 128 + col] + v;
      }
    }
  }
}

// ---------------------------------------------------------------------------
// Node MLP kernel: one wave per 32 nodes; in = [recv, aggr]; out = recv + n_up.
// ---------------------------------------------------------------------------
__global__ __launch_bounds__(64)
void node_mlp_kernel(const float* __restrict__ recv,
                     const float* __restrict__ aggr,
                     const unsigned short* __restrict__ W1p,
                     const unsigned short* __restrict__ W2p,
                     const float* __restrict__ b1, const float* __restrict__ b2,
                     const float* __restrict__ ls, const float* __restrict__ lb,
                     float* __restrict__ out, int N)
{
  constexpr int KIN = 256, XP = KIN + LDPAD;
  __shared__ unsigned short smem[2 * 32 * XP];          // 33792 B
  const int wid  = threadIdx.x >> 5;
  const int lane = threadIdx.x & 31;
  const int lg = lane >> 4, ln = lane & 15;
  unsigned short* xs = smem + (unsigned)wid * 32 * XP;
  const int n0 = (blockIdx.x * 2 + wid) * 32;
  if (n0 >= N) return;

  const int c = lane * 4;
#pragma unroll 1
  for (int m = 0; m < 32; ++m) {
    int n = n0 + m;
    st4bf(xs + m * XP +   0 + c, ld4(recv + (size_t)n * 128 + c));
    st4bf(xs + m * XP + 128 + c, ld4(aggr + (size_t)n * 128 + c));
  }

  v8f acc2[2][8];
  wave_mlp2_r32<256>(xs, W1p, W2p, b1, b2, acc2, lane);
  wave_ln(acc2[0], ls, lb, lane);
  wave_ln(acc2[1], ls, lb, lane);

#pragma unroll
  for (int st = 0; st < 2; ++st) {
#pragma unroll
    for (int r = 0; r < 8; ++r) {
      int n = n0 + st * 16 + r + lg * 8;
#pragma unroll
      for (int nt = 0; nt < 8; ++nt) {
        int col = nt * 16 + ln;
        out[(size_t)n * 128 + col] = recv[(size_t)n * 128 + col] + acc2[st][nt][r];
      }
    }
  }
}

// ---------------------------------------------------------------------------
// Weight pre-pack: f32 row-major [G][Ktot][128] -> bf16 WMMA B-fragments
// out[(((g*KT+kt)*8+nt)*32+lane)*16+j] = W[g][kt*32+(lane>>4)*8+(j&7)+(j>>3)*16][nt*16+(lane&15)]
// ---------------------------------------------------------------------------
__global__ void pack_w_kernel(const float* __restrict__ W,
                              unsigned short* __restrict__ out,
                              int Ktot, long long total)
{
  long long t = (long long)blockIdx.x * blockDim.x + threadIdx.x;
  if (t >= total) return;
  int j    = (int)(t & 15);
  int lane = (int)((t >> 4) & 31);
  int nt   = (int)((t >> 9) & 7);
  long long rest = t >> 12;
  int KT = Ktot >> 5;
  int kt = (int)(rest % KT);
  long long g = rest / KT;
  int K = kt * 32 + (lane >> 4) * 8 + (j & 7) + (j >> 3) * 16;
  int N = nt * 16 + (lane & 15);
  out[t] = f2bfbits(W[(g * Ktot + K) * 128 + N]);
}

__global__ void zero_kernel(float* __restrict__ p, long long n) {
  long long i = (long long)blockIdx.x * blockDim.x + threadIdx.x;
  if (i < n) p[i] = 0.f;
}

__global__ void add_kernel(float* __restrict__ c, const float* __restrict__ a,
                           const float* __restrict__ b, long long n) {
  long long i = (long long)blockIdx.x * blockDim.x + threadIdx.x;
  if (i < n) c[i] = a[i] + b[i];
}

// ---------------------------------------------------------------------------
// Tiny decoder head: 1024 rows, 128->silu->64; mu / std split.  Negligible
// FLOPs -> plain VALU kernel, one row per 64-thread block.
// ---------------------------------------------------------------------------
__global__ __launch_bounds__(64)
void head_kernel(const float* __restrict__ rep,
                 const float* __restrict__ W1, const float* __restrict__ b1,
                 const float* __restrict__ W2, const float* __restrict__ b2,
                 float* __restrict__ mu, float* __restrict__ stdo)
{
  __shared__ float xr[128];
  __shared__ float hb[128];
  const int row = blockIdx.x;
  const int t = threadIdx.x;
  xr[t]      = rep[(size_t)row * 128 + t];
  xr[t + 64] = rep[(size_t)row * 128 + t + 64];
  __syncthreads();
#pragma unroll
  for (int jj = 0; jj < 2; ++jj) {
    int j = t + jj * 64;
    float s = b1[j];
    for (int k = 0; k < 128; ++k) s += xr[k] * W1[k * 128 + j];
    hb[j] = s / (1.f + __expf(-s));
  }
  __syncthreads();
  float s = b2[t];
  for (int k = 0; k < 128; ++k) s += hb[k] * W2[k * 64 + t];
  if (t < 32) {
    mu[(size_t)row * 32 + t] = s;
  } else {
    float sp = (s > 20.f) ? s : log1pf(__expf(s));
    stdo[(size_t)row * 32 + (t - 32)] = 1e-4f + sp;
  }
}

// ---------------------------------------------------------------------------
// Host orchestration
// ---------------------------------------------------------------------------
extern "C" void kernel_launch(void* const* d_in, const int* in_sizes, int n_in,
                              void* d_out, int out_size, void* d_ws, size_t ws_size,
                              hipStream_t stream)
{
  (void)in_sizes; (void)n_in; (void)out_size; (void)ws_size;
  constexpr int H = 128;
  constexpr int S1 = 16384, S2 = 4096, S3 = 1024;

  const float* high_emb = (const float*)d_in[0];
  const float* low_emb  = (const float*)d_in[1];
  const float* hr_mesh0 = (const float*)d_in[2];
  const float* hr_mesh1 = (const float*)d_in[3];
  const float* hr_mesh2 = (const float*)d_in[4];
  const float* lr_mesh0 = (const float*)d_in[5];
  const float* lr_mesh1 = (const float*)d_in[6];

  struct Edge { const float* feat; const int* src; const int* dst; int E; };
  auto mkE = [&](int base, int E) {
    return Edge{(const float*)d_in[base], (const int*)d_in[base+1], (const int*)d_in[base+2], E};
  };
  Edge hr_g2m  = mkE(7, 131072);
  Edge hr_m2m0 = mkE(10, 131072);
  Edge hr_m2m1 = mkE(13, 32768);
  Edge hr_m2m2 = mkE(16, 8192);
  Edge hr_up0  = mkE(19, 65536);
  Edge hr_up1  = mkE(22, 16384);
  Edge lr_g2m  = mkE(25, 65536);
  Edge lr_m2m0 = mkE(28, 32768);
  Edge lr_m2m1 = mkE(31, 8192);
  Edge lr_up0  = mkE(34, 16384);

  const float* eW1 = (const float*)d_in[37];
  const float* eb1 = (const float*)d_in[38];
  const float* eW2 = (const float*)d_in[39];
  const float* eb2 = (const float*)d_in[40];
  const float* elns = (const float*)d_in[41];
  const float* elnb = (const float*)d_in[42];
  const float* nW1 = (const float*)d_in[43];
  const float* nb1 = (const float*)d_in[44];
  const float* nW2 = (const float*)d_in[45];
  const float* nb2 = (const float*)d_in[46];
  const float* nlns = (const float*)d_in[47];
  const float* nlnb = (const float*)d_in[48];
  const float* pW1 = (const float*)d_in[49];
  const float* pb1 = (const float*)d_in[50];
  const float* pW2 = (const float*)d_in[51];
  const float* pb2 = (const float*)d_in[52];

  // ---- workspace carve (~140 MB) ----
  char* w = (char*)d_ws;
  unsigned short* eW1p = (unsigned short*)w; w += (size_t)15 * 12 * 4096 * 2;
  unsigned short* eW2p = (unsigned short*)w; w += (size_t)15 *  4 * 4096 * 2;
  unsigned short* nW1p = (unsigned short*)w; w += (size_t)15 *  8 * 4096 * 2;
  unsigned short* nW2p = (unsigned short*)w; w += (size_t)15 *  4 * 4096 * 2;
  float* aggr = (float*)w; w += (size_t)S1 * H * 4;
  float* ebuf = (float*)w; w += (size_t)131072 * H * 4;
  float* A = (float*)w; w += (size_t)S1 * H * 4;
  float* B = (float*)w; w += (size_t)S1 * H * 4;
  float* C = (float*)w; w += (size_t)S1 * H * 4;
  float* D = (float*)w; w += (size_t)S1 * H * 4;
  float* Ebuf2 = (float*)w; w += (size_t)S1 * H * 4;
  float* F = (float*)w; w += (size_t)S1 * H * 4;

  // ---- output layout: mu, std, low_emb, lr_in0, lr_in1, lr_up_rep0, lr_up_rep1
  float* out_mu     = (float*)d_out;
  float* out_std    = out_mu  + 1024 * 32;
  float* out_low    = out_std + 1024 * 32;
  float* out_lrin0  = out_low   + (size_t)S1 * H;
  float* out_lrin1  = out_lrin0 + (size_t)S2 * H;
  float* out_lr0    = out_lrin1 + (size_t)S3 * H;
  float* out_lr1    = out_lr0   + (size_t)S2 * H;

  // ---- pack weights to bf16 WMMA B-fragment layout (once per launch) ----
  {
    long long t1 = 15LL * 12 * 4096, t2 = 15LL * 4 * 4096, t3 = 15LL * 8 * 4096;
    pack_w_kernel<<<(unsigned)((t1 + 255) / 256), 256, 0, stream>>>(eW1, eW1p, 384, t1);
    pack_w_kernel<<<(unsigned)((t2 + 255) / 256), 256, 0, stream>>>(eW2, eW2p, 128, t2);
    pack_w_kernel<<<(unsigned)((t3 + 255) / 256), 256, 0, stream>>>(nW1, nW1p, 256, t3);
    pack_w_kernel<<<(unsigned)((t2 + 255) / 256), 256, 0, stream>>>(nW2, nW2p, 128, t2);
  }

  auto zero = [&](float* p, long long n) {
    zero_kernel<<<(unsigned)((n + 511) / 512), 512, 0, stream>>>(p, n);
  };
  auto addv = [&](float* c, const float* a, const float* b, long long n) {
    add_kernel<<<(unsigned)((n + 511) / 512), 512, 0, stream>>>(c, a, b, n);
  };
  // One full GNN layer: edge MLP + scatter-sum + node MLP.
  // 64-thread blocks = 2 waves; each wave owns 32 rows -> block covers 64.
  auto gnn = [&](int g, const float* feat, const int* esrc, const int* edst, int E,
                 const float* send, const float* recv, int Nrecv,
                 float* nodeOut, float* edgeOut) {
    zero(aggr, (long long)Nrecv * H);
    edge_mlp_kernel<<<E / 64, 64, 0, stream>>>(
        feat, esrc, edst, send, recv,
        eW1p + (size_t)g * 12 * 4096, eW2p + (size_t)g * 4 * 4096,
        eb1 + g * H, eb2 + g * H, elns + g * H, elnb + g * H,
        aggr, edgeOut, E);
    node_mlp_kernel<<<Nrecv / 64, 64, 0, stream>>>(
        recv, aggr,
        nW1p + (size_t)g * 8 * 4096, nW2p + (size_t)g * 4 * 4096,
        nb1 + g * H, nb2 + g * H, nlns + g * H, nlnb + g * H,
        nodeOut, Nrecv);
  };

  // ---------------- dataflow (matches reference) ----------------
  // gnn0: high grid -> hr_mesh0
  gnn(0, hr_g2m.feat, hr_g2m.src, hr_g2m.dst, hr_g2m.E, high_emb, hr_mesh0, S1, A, nullptr);
  // intra(3,4) on A with hr_m2m0 (edge residual via ebuf)
  gnn(3, hr_m2m0.feat, hr_m2m0.src, hr_m2m0.dst, hr_m2m0.E, A, A, S1, B, ebuf);
  gnn(4, ebuf,         hr_m2m0.src, hr_m2m0.dst, hr_m2m0.E, B, B, S1, A, nullptr);
  addv(C, A, low_emb, (long long)S1 * H);                 // rep = hr_in + low_emb
  // gnn1: rep -> hr_mesh1
  gnn(1, hr_up0.feat, hr_up0.src, hr_up0.dst, hr_up0.E, C, hr_mesh1, S2, D, nullptr);
  // gnn9: low_emb -> lr_mesh0  (lr_up_rep0 is a final output)
  gnn(9, lr_g2m.feat, lr_g2m.src, lr_g2m.dst, lr_g2m.E, low_emb, lr_mesh0, S2, out_lr0, nullptr);
  addv(Ebuf2, D, out_lr0, (long long)S2 * H);             // rep_s2
  // intra(5,6)
  gnn(5, hr_m2m1.feat, hr_m2m1.src, hr_m2m1.dst, hr_m2m1.E, Ebuf2, Ebuf2, S2, F, ebuf);
  gnn(6, ebuf,         hr_m2m1.src, hr_m2m1.dst, hr_m2m1.E, F, F, S2, Ebuf2, nullptr);  // hr_in_s2
  // intra(11,12) on lr_up_rep0 -> lr_in0 (final output)
  gnn(11, lr_m2m0.feat, lr_m2m0.src, lr_m2m0.dst, lr_m2m0.E, out_lr0, out_lr0, S2, F, ebuf);
  gnn(12, ebuf,         lr_m2m0.src, lr_m2m0.dst, lr_m2m0.E, F, F, S2, out_lrin0, nullptr);
  addv(D, Ebuf2, out_lrin0, (long long)S2 * H);           // rep
  // gnn2: rep -> hr_mesh2
  gnn(2, hr_up1.feat, hr_up1.src, hr_up1.dst, hr_up1.E, D, hr_mesh2, S3, A, nullptr);
  // gnn10: lr_in0 -> lr_mesh1 (lr_up_rep1 is a final output)
  gnn(10, lr_up0.feat, lr_up0.src, lr_up0.dst, lr_up0.E, out_lrin0, lr_mesh1, S3, out_lr1, nullptr);
  addv(B, A, out_lr1, (long long)S3 * H);                 // rep_s3
  // intra(7,8)
  gnn(7, hr_m2m2.feat, hr_m2m2.src, hr_m2m2.dst, hr_m2m2.E, B, B, S3, C, ebuf);
  gnn(8, ebuf,         hr_m2m2.src, hr_m2m2.dst, hr_m2m2.E, C, C, S3, B, nullptr);      // hr_in_s3
  // intra(13,14) on lr_up_rep1 -> lr_in1 (final output)
  gnn(13, lr_m2m1.feat, lr_m2m1.src, lr_m2m1.dst, lr_m2m1.E, out_lr1, out_lr1, S3, C, ebuf);
  gnn(14, ebuf,         lr_m2m1.src, lr_m2m1.dst, lr_m2m1.E, C, C, S3, out_lrin1, nullptr);
  addv(D, B, out_lrin1, (long long)S3 * H);               // final rep
  // head -> mu, std
  head_kernel<<<S3, 64, 0, stream>>>(D, pW1, pb1, pW2, pb2, out_mu, out_std);
  // low_emb passthrough output
  hipMemcpyAsync(out_low, low_emb, (size_t)S1 * H * sizeof(float),
                 hipMemcpyDeviceToDevice, stream);
}
